// WindowsMultiheadAttention_89610197663925
// MI455X (gfx1250) — matrix operand order repaired and checked by reading, not compiled, not run on hardware
//
#include <hip/hip_runtime.h>
#include <cstdint>
#include <cstddef>

// CDNA5 / gfx1250, wave32. One workgroup (256 threads = 8 waves) per 7x7 window.
// All GEMMs via v_wmma_f32_16x16x32_f16; weight tiles double-buffered into LDS
// with GLOBAL_LOAD_ASYNC_TO_LDS_B128 (ASYNCcnt) when available.

typedef __attribute__((ext_vector_type(16))) _Float16 v16h;
typedef __attribute__((ext_vector_type(8)))  _Float16 v8h;
typedef __attribute__((ext_vector_type(4)))  _Float16 v4h;
typedef __attribute__((ext_vector_type(8)))  float    v8f;
typedef __attribute__((ext_vector_type(4)))  float    v4f;
typedef int v4i __attribute__((vector_size(16)));     // builtin wants int4* (generic AS)

#define WSTRIDE   264             // winA / o / Wstage row stride (halves): 256 + 8 pad
#define QSTRIDE   776             // qkv row stride (halves): 768 + 8 pad
#define VTSTRIDE  72              // vT row stride (halves): 64 + 8 pad
#define SSTRIDE   68              // scores row stride (floats): 64 + 4 pad
#define SHEAD     (64 * SSTRIDE)  // floats per head score block
#define ASTRIDEH  (SSTRIDE * 2)   // score row stride in halves (f16 attn reuses buffer)
#define WBUFH     (64 * WSTRIDE)  // halves per weight staging buffer (33792 B)

// LDS layout (dynamic shared, ~319KB total, <= 320KB per workgroup):
#define OFF_SCORES   0            // 8*64*68*4 = 139264 B (union w/ 2x Wstage 64x264 f16)
#define OFF_AMEAN    139264       // 49*50*4 -> pad 9808 B
#define OFF_WINA     149072       // 64*264*2 =  33792 B (reused as o-matrix)
#define OFF_QKV      182864       // 64*776*2 =  99328 B
#define OFF_VT       282192       // 256*72*2 =  36864 B
#define SMEM_BYTES   319056

#if __has_builtin(__builtin_amdgcn_global_load_async_to_lds_b128) && \
    __has_builtin(__builtin_amdgcn_s_wait_asynccnt)
#define HAVE_ASYNC 1
#else
#define HAVE_ASYNC 0
#endif

__device__ __forceinline__ void async_cp16(const _Float16* gsrc, _Float16* ldst) {
#if HAVE_ASYNC
  __builtin_amdgcn_global_load_async_to_lds_b128((v4i*)gsrc, (v4i*)ldst, 0, 0);
#else
  *(v8h*)ldst = *(const v8h*)gsrc;
#endif
}

template <int N>
__device__ __forceinline__ void wait_async() {
#if HAVE_ASYNC
  __builtin_amdgcn_s_wait_asynccnt(N);
#endif
}

// A fragment (16x32 f16): lane = m (mod 16); halves j<8 -> K = kb + j, j>=8 -> K = kb+16+j-8,
// kb = kc*32 + (lane>=16 ? 8 : 0). Two 16B LDS loads.
__device__ __forceinline__ v16h frag_a(const _Float16* p) {
  v8h l0 = *(const v8h*)p;
  v8h l1 = *(const v8h*)(p + 16);
  v16h r;
#pragma unroll
  for (int i = 0; i < 8; ++i) { r[i] = l0[i]; r[i + 8] = l1[i]; }
  return r;
}
// B fragment (32x16 f16): lane = n (mod 16); halves j<8 -> K = kb + j, j>=8 -> K = kb+16+j-8,
// kb = kc*32 + (lane>=16 ? 8 : 0).
__device__ __forceinline__ v16h frag_b(const _Float16* p) {
  v8h l0 = *(const v8h*)p;
  v8h l1 = *(const v8h*)(p + 16);
  v16h r;
#pragma unroll
  for (int i = 0; i < 8; ++i) { r[i] = l0[i]; r[i + 8] = l1[i]; }
  return r;
}

#define WMMA_F16(A, B, C) \
  __builtin_amdgcn_wmma_f32_16x16x32_f16(false, (A), false, (B), (short)0, (C), false, false)

// One-time f32 -> f16 conversion of both weight matrices into workspace.
__global__ __launch_bounds__(256)
void cvt_weights(const float* __restrict__ wq, const float* __restrict__ wo,
                 _Float16* __restrict__ out) {
  int i = blockIdx.x * 256 + threadIdx.x;
  if (i < 196608)       out[i] = (_Float16)wq[i];
  else if (i < 262144)  out[i] = (_Float16)wo[i - 196608];
}

__global__ __launch_bounds__(256, 1)
void win_mha_kernel(const float* __restrict__ x,      // (64, 3136, 256)
                    const float* __restrict__ wqkv,   // (768, 256) f32
                    const float* __restrict__ bqkv,   // (768,)
                    const float* __restrict__ wout,   // (256, 256) f32
                    const float* __restrict__ bout,   // (256,)
                    const _Float16* __restrict__ wh,  // f16 weights in ws (qkv | out)
                    int use_ws,
                    float* __restrict__ y,            // (4096, 49, 256)
                    float* __restrict__ attn_out)     // (4096, 49, 49)
{
  extern __shared__ __align__(16) char smem[];
  float*    scores   = (float*)(smem + OFF_SCORES);
  _Float16* Wstage   = (_Float16*)(smem + OFF_SCORES);  // 2 buffers, union w/ scores
  float*    attnMean = (float*)(smem + OFF_AMEAN);
  _Float16* winA     = (_Float16*)(smem + OFF_WINA);    // window f16; later o-matrix
  _Float16* qkv      = (_Float16*)(smem + OFF_QKV);     // (64, 768) f16: q|k|v
  _Float16* vT       = (_Float16*)(smem + OFF_VT);      // (256, 64) f16: v transposed

  const int tid  = threadIdx.x;
  const int wv   = tid >> 5;        // wave id 0..7 (wave32)
  const int lane = tid & 31;
  const int ln   = lane & 15;
  const int hi   = lane >> 4;       // half-wave select

  const int mwin = blockIdx.x;      // window index 0..4095
  const int b    = mwin >> 6;
  const int ww   = mwin & 63;
  const int hk   = ww >> 3, wk = ww & 7;
  const float* xb = x + (size_t)b * 3136 * 256;

  const _Float16* wqkv_h = wh;            // (768,256) f16
  const _Float16* wout_h = wh + 196608;   // (256,256) f16

  // Stage one 64-row weight group into staging buffer `bufi`.
  // srcH: f16 rows (group base) in ws; srcF: f32 rows (group base) in global.
  auto stage = [&](const _Float16* srcH, const float* srcF, int bufi) {
    _Float16* dst = Wstage + bufi * WBUFH;
    if (use_ws) {
#pragma unroll
      for (int rep = 0; rep < 8; ++rep) {       // 2048 chunks of 8 halves (16 B)
        int chunk = rep * 256 + tid;
        int c = chunk >> 5, k8 = (chunk & 31) * 8;
        async_cp16(srcH + c * 256 + k8, dst + c * WSTRIDE + k8);
      }
    } else {
#pragma unroll 1
      for (int rep = 0; rep < 16; ++rep) {      // f32 -> f16 convert path
        int chunk = rep * 256 + tid;
        int c = chunk >> 6, k4 = (chunk & 63) * 4;
        v4f w4 = *(const v4f*)(srcF + (size_t)c * 256 + k4);
        v4h h4;
#pragma unroll
        for (int i = 0; i < 4; ++i) h4[i] = (_Float16)w4[i];
        *(v4h*)(dst + c * WSTRIDE + k4) = h4;
      }
    }
  };

  // ---- Prologue: start async copy of qkv weight group 0 (overlaps the gather)
  stage(wqkv_h, wqkv, 0);

  // ---- Phase 1: gather scrambled window -> winA (f16); zero pads; zero attnMean
  // win[l][e] with i = l*256 + e, i = ch*49 + s; thread = ch (coalesced x reads).
#pragma unroll 1
  for (int s = 0; s < 49; ++s) {
    int n = (hk * 7 + s / 7) * 56 + wk * 7 + (s % 7);
    int i = tid * 49 + s;
    winA[(i >> 8) * WSTRIDE + (i & 255)] = (_Float16)xb[(size_t)n * 256 + tid];
  }
  for (int i = tid; i < 15 * WSTRIDE; i += 256) winA[49 * WSTRIDE + i] = (_Float16)0.f;
  for (int i = tid; i < 49 * 50; i += 256) attnMean[i] = 0.f;

  const float QSCALE = 0.17677669529663687f;  // 1/sqrt(32), folded into q
  const int rtA = (wv >> 2) * 2;
  const int rtB = rtA + 1;
  const int ntl = wv & 3;

  // ---- Phase 2: qkv = win @ Wqkv^T + b. 12 groups of 64 cols, double-buffered.
#pragma unroll 1
  for (int g = 0; g < 12; ++g) {
    if (g + 1 < 12) {
      stage(wqkv_h + (size_t)(g + 1) * 64 * 256, wqkv + (size_t)(g + 1) * 64 * 256,
            (g + 1) & 1);
      if (use_ws) wait_async<8>();   // group g's 8 per-wave async ops retired
    } else {
      if (use_ws) wait_async<0>();
    }
    __syncthreads();

    const _Float16* cur = Wstage + (g & 1) * WBUFH;
    v8f accA = {}; v8f accB = {};
#pragma unroll
    for (int kc = 0; kc < 8; ++kc) {
      v16h bf = frag_b(cur + (ntl * 16 + ln) * WSTRIDE + kc * 32 + hi * 8);
      v16h aA = frag_a(winA + (rtA * 16 + ln) * WSTRIDE + kc * 32 + hi * 8);
      accA = WMMA_F16(aA, bf, accA);
      v16h aB = frag_a(winA + (rtB * 16 + ln) * WSTRIDE + kc * 32 + hi * 8);
      accB = WMMA_F16(aB, bf, accB);
    }
    int colg   = g * 64 + ntl * 16 + ln;
    float bias = bqkv[colg];
    float scl  = (colg < 256) ? QSCALE : 1.f;
#pragma unroll
    for (int r = 0; r < 8; ++r) {
      int mA = rtA * 16 + r + hi * 8;
      int mB = rtB * 16 + r + hi * 8;
      float vA = (accA[r] + bias) * scl;
      float vB = (accB[r] + bias) * scl;
      qkv[mA * QSTRIDE + colg] = (_Float16)vA;
      qkv[mB * QSTRIDE + colg] = (_Float16)vB;
      if (colg >= 512) {                      // v: also store transposed for attn@v B-frags
        vT[(colg - 512) * VTSTRIDE + mA] = (_Float16)vA;
        vT[(colg - 512) * VTSTRIDE + mB] = (_Float16)vB;
      }
    }
    __syncthreads();   // protects buf (g&1) before it is re-issued at iteration g+1
  }

  // ---- Phase 3: scores[h] = q_h @ k_h^T  (wave = head; K=32 -> one WMMA per tile)
  {
    float* sh = scores + wv * SHEAD;
#pragma unroll
    for (int rt = 0; rt < 4; ++rt) {
      v16h aq = frag_a(qkv + (rt * 16 + ln) * QSTRIDE + wv * 32 + hi * 8);
#pragma unroll
      for (int nt = 0; nt < 4; ++nt) {
        v16h bk = frag_b(qkv + (nt * 16 + ln) * QSTRIDE + 256 + wv * 32 + hi * 8);
        v8f acc = {};
        acc = WMMA_F16(aq, bk, acc);
#pragma unroll
        for (int r = 0; r < 8; ++r)
          sh[(rt * 16 + r + hi * 8) * SSTRIDE + nt * 16 + ln] = acc[r];
      }
    }
  }
  __syncthreads();

  // ---- Phase 4: softmax over valid 49 cols; write attn f16 in place; mean over heads
  {
    float*    sh = scores + wv * SHEAD;
    _Float16* ah = (_Float16*)sh;
#pragma unroll 1
    for (int pass = 0; pass < 2; ++pass) {
      int r = lane + pass * 32;
      if (r < 49) {
        float vals[49];
        float mx = -3.0e38f;
#pragma unroll
        for (int c = 0; c < 49; ++c) { vals[c] = sh[r * SSTRIDE + c]; mx = fmaxf(mx, vals[c]); }
        float sum = 0.f;
#pragma unroll
        for (int c = 0; c < 49; ++c) { float e = __expf(vals[c] - mx); vals[c] = e; sum += e; }
        float inv = 1.f / sum;
        _Float16* arow = ah + r * ASTRIDEH;
#pragma unroll
        for (int c = 0; c < 49; ++c) {
          float a = vals[c] * inv;
          arow[c] = (_Float16)a;
          atomicAdd(&attnMean[r * 50 + c], a * 0.125f);   // ds_add_f32
        }
#pragma unroll
        for (int c = 49; c < 64; ++c) arow[c] = (_Float16)0.f;  // mask pad K
      } else if (pass == 1 && r < 64) {
        _Float16* arow = ah + r * ASTRIDEH;                 // zero pad rows
#pragma unroll
        for (int c = 0; c < 64; ++c) arow[c] = (_Float16)0.f;
      }
    }
  }
  __syncthreads();

  // write head-averaged attention map
  for (int i = tid; i < 49 * 49; i += 256)
    attn_out[(size_t)mwin * 2401 + i] = attnMean[(i / 49) * 50 + (i % 49)];

  // ---- Phase 5: o_h = attn_h @ v_h  (wave = head; K=64 padded, 2 chunks), o -> winA
  {
    const _Float16* ah = (const _Float16*)(scores + wv * SHEAD);
#pragma unroll
    for (int rt = 0; rt < 4; ++rt) {
#pragma unroll
      for (int dt = 0; dt < 2; ++dt) {
        v8f acc = {};
#pragma unroll
        for (int kc = 0; kc < 2; ++kc) {
          v16h af = frag_a(ah + (rt * 16 + ln) * ASTRIDEH + kc * 32 + hi * 8);
          v16h bf = frag_b(vT + (wv * 32 + dt * 16 + ln) * VTSTRIDE + kc * 32 + hi * 8);
          acc = WMMA_F16(af, bf, acc);
        }
#pragma unroll
        for (int r = 0; r < 8; ++r)
          winA[(rt * 16 + r + hi * 8) * WSTRIDE + wv * 32 + dt * 16 + ln] = (_Float16)acc[r];
      }
    }
  }
  __syncthreads();   // attn region now dead -> staging buffers may reuse it

  // ---- Phase 6: y = o @ Wout^T + b. 4 groups of 64 cols, double-buffered.
  stage(wout_h, wout, 0);
#pragma unroll 1
  for (int g = 0; g < 4; ++g) {
    if (g + 1 < 4) {
      stage(wout_h + (size_t)(g + 1) * 64 * 256, wout + (size_t)(g + 1) * 64 * 256,
            (g + 1) & 1);
      if (use_ws) wait_async<8>();
    } else {
      if (use_ws) wait_async<0>();
    }
    __syncthreads();

    const _Float16* cur = Wstage + (g & 1) * WBUFH;
    v8f accA = {}; v8f accB = {};
#pragma unroll
    for (int kc = 0; kc < 8; ++kc) {
      v16h bf = frag_b(cur + (ntl * 16 + ln) * WSTRIDE + kc * 32 + hi * 8);
      v16h aA = frag_a(winA + (rtA * 16 + ln) * WSTRIDE + kc * 32 + hi * 8);
      accA = WMMA_F16(aA, bf, accA);
      v16h aB = frag_a(winA + (rtB * 16 + ln) * WSTRIDE + kc * 32 + hi * 8);
      accB = WMMA_F16(aB, bf, accB);
    }
    int col    = g * 64 + ntl * 16 + ln;
    float bias = bout[col];
#pragma unroll
    for (int r = 0; r < 8; ++r) {
      int mA = rtA * 16 + r + hi * 8;
      if (mA < 49) y[((size_t)mwin * 49 + mA) * 256 + col] = accA[r] + bias;
      int mB = rtB * 16 + r + hi * 8;
      if (mB < 49) y[((size_t)mwin * 49 + mB) * 256 + col] = accB[r] + bias;
    }
    __syncthreads();
  }
}

extern "C" void kernel_launch(void* const* d_in, const int* in_sizes, int n_in,
                              void* d_out, int out_size, void* d_ws, size_t ws_size,
                              hipStream_t stream) {
  (void)in_sizes; (void)n_in; (void)out_size;
  const float* x    = (const float*)d_in[0];
  const float* wqkv = (const float*)d_in[1];
  const float* bqkv = (const float*)d_in[2];
  const float* wout = (const float*)d_in[3];
  const float* bout = (const float*)d_in[4];
  float* y    = (float*)d_out;
  float* attn = y + (size_t)4096 * 49 * 256;  // second tuple output, concatenated flat

  const int use_ws = (ws_size >= (size_t)262144 * sizeof(_Float16)) ? 1 : 0;
  if (use_ws)
    cvt_weights<<<1024, 256, 0, stream>>>(wqkv, wout, (_Float16*)d_ws);

  (void)hipFuncSetAttribute((const void*)win_mha_kernel,
                            hipFuncAttributeMaxDynamicSharedMemorySize, SMEM_BYTES);
  win_mha_kernel<<<4096, 256, SMEM_BYTES, stream>>>(
      x, wqkv, bqkv, wout, bout, (const _Float16*)d_ws, use_ws, y, attn);
}
